// SOAP_pred_head_nenv_52888227283048
// MI455X (gfx1250) — matrix operand order repaired and checked by reading, not compiled, not run on hardware
//
#include <hip/hip_runtime.h>

// CDNA5 / gfx1250, wave32. All matmuls via v_wmma_f32_16x16x32_f16.
typedef __attribute__((ext_vector_type(16))) _Float16 v16h;
typedef __attribute__((ext_vector_type(8)))  _Float16 v8h;
typedef __attribute__((ext_vector_type(8)))  float    v8f;

#define WMMA_F16(acc, a, b) \
  (acc) = __builtin_amdgcn_wmma_f32_16x16x32_f16(false, (a), false, (b), (short)0, (acc), false, false)

struct Params {
  const float* Lnce; const float* Hnce; const float* nL; const float* nH;
  const float* a1[3][4];   // [q,k,v][W1,W2,b1,b2]
  const float* a2[3][4];
  const float* a3[3][4];
  const float* sc[4];      // aggr score W1,W2,b1,b2
  const float* vl[4];      // aggr val   W1,W2,b1,b2
  const float* ln1g[2]; const float* ln1b[2];   // [L,H], dim 32
  const float* ln2g[2]; const float* ln2b[2];   // dim 16
  const float* ln3g[2]; const float* ln3b[2];   // dim 16
};

// f16 weight workspace offsets. Weights are stored FRAGMENT-SWIZZLED:
// per 32x16 (KxN) tile, 512 f16 laid out as [lane(32)][j(16)] with
// element j of lane (lhi*16+lr) = W[kb + lhi*16 + j][nt + lr].
#define OFF_A1W1 0       // 3 * (96x32  -> 6 tiles * 512)
#define OFF_A1W2 9216    // 3 * (32x32  -> 2 tiles)
#define OFF_A2W1 12288   // 3 * (64x16  -> 2 tiles)
#define OFF_A2W2 15360   // 3 * (32x16  -> 1 tile)
#define OFF_A3W1 16896   // 3 * (32x16)
#define OFF_A3W2 18432   // 3 * (32x16)
#define OFF_SCW1 19968   // 32x16
#define OFF_SCW2 20480   // 32x16 (N 1 -> 16)
#define OFF_VLW1 20992   // 32x16
#define OFF_VLW2 21504   // 32x16

// LDS layout (bytes)
#define SM_QKV   24576   // q/k f16 [128][32]; v stored transposed f16 [32][128]
#define SM_HS    49152   // hidden / scores, 8KB
#define SM_O2    57344   // out2 f32 [128][16], 8KB

// ---------------- weight prep: f32 -> fragment-swizzled f16 in d_ws ----------------
__global__ void soap_prep(Params P, _Float16* __restrict__ ws) {
  int m = blockIdx.x;
  const float* src; int K, N, Kp, off;
  if      (m < 3)  { src = P.a1[m][0];    K = 72; N = 32; Kp = 96; off = OFF_A1W1 + m * 3072; }
  else if (m < 6)  { src = P.a1[m-3][1];  K = 32; N = 32; Kp = 32; off = OFF_A1W2 + (m-3) * 1024; }
  else if (m < 9)  { src = P.a2[m-6][0];  K = 40; N = 16; Kp = 64; off = OFF_A2W1 + (m-6) * 1024; }
  else if (m < 12) { src = P.a2[m-9][1];  K = 16; N = 16; Kp = 32; off = OFF_A2W2 + (m-9) * 512; }
  else if (m < 15) { src = P.a3[m-12][0]; K = 24; N = 16; Kp = 32; off = OFF_A3W1 + (m-12) * 512; }
  else if (m < 18) { src = P.a3[m-15][1]; K = 16; N = 16; Kp = 32; off = OFF_A3W2 + (m-15) * 512; }
  else if (m == 18){ src = P.sc[0];       K = 16; N = 16; Kp = 32; off = OFF_SCW1; }
  else if (m == 19){ src = P.sc[1];       K = 16; N = 1;  Kp = 32; off = OFF_SCW2; }
  else if (m == 20){ src = P.vl[0];       K = 16; N = 16; Kp = 32; off = OFF_VLW1; }
  else             { src = P.vl[1];       K = 16; N = 16; Kp = 32; off = OFF_VLW2; }
  int Np = (m < 6) ? 32 : 16;
  int ntn = Np >> 4;               // N tiles
  int total = Kp * Np;
  for (int i = threadIdx.x; i < total; i += blockDim.x) {
    int tile = i >> 9;             // 512 f16 per tile
    int lane = (i >> 4) & 31;
    int j    = i & 15;
    int kt   = tile / ntn, nt = tile % ntn;
    int k = kt * 32 + (lane >> 4) * 16 + j;
    int n = nt * 16 + (lane & 15);
    float v = (k < K && n < N) ? src[k * N + n] : 0.f;
    ws[off + i] = (_Float16)v;
  }
}

// ---------------- vector fragment loads ----------------
// 16 f16 = [p[0..7], p[16..23]]  (A-matrix row layout, 16B aligned)
__device__ __forceinline__ v16h ld_gap(const _Float16* p) {
  v8h lo = *(const v8h*)p;
  v8h hi = *(const v8h*)(p + 16);
  return __builtin_shufflevector(lo, hi, 0,1,2,3,4,5,6,7,8,9,10,11,12,13,14,15);
}
// 16 contiguous f16 (B-matrix fragments, 16B aligned)
__device__ __forceinline__ v16h ld_cat(const _Float16* p) {
  v8h lo = *(const v8h*)p;
  v8h hi = *(const v8h*)(p + 8);
  return __builtin_shufflevector(lo, hi, 0,1,2,3,4,5,6,7,8,9,10,11,12,13,14,15);
}

// A-fragment (16x32 f16) from row-major [rows][stride], contiguous k = 16*lhi + j
__device__ __forceinline__ v16h frag_A(const _Float16* base, int stride, int row0,
                                       int kb, int lr, int lhi) {
  return ld_cat(base + (row0 + lr) * stride + kb + lhi * 16);
}
// B-fragment from fragment-swizzled weight buffer (Np = padded N of that weight)
__device__ __forceinline__ v16h frag_W(const _Float16* W, int Np, int kb, int nt, int lane) {
  return ld_cat(W + (((kb >> 5) * (Np >> 4) + (nt >> 4)) << 9) + lane * 16);
}

// Y = act(X[128][xstride(Kp)] @ W_swizzled[Kp][N] + bias), this wave's 16 rows.
// toVT: write transposed into VT[32][128] (for attention V); else row-major [128][ystride].
__device__ __forceinline__ void mlp_f16(const _Float16* Xin, int xstride, int Kp,
                                        const _Float16* W, const float* bias, int N,
                                        _Float16* Yout, int ystride, int zero_to,
                                        bool relu, bool toVT,
                                        int row0, int lr, int lhi, int lane) {
  for (int nt = 0; nt < N; nt += 16) {
    v8f acc = {};
    for (int kb = 0; kb < Kp; kb += 32) {
      v16h a = frag_A(Xin, xstride, row0, kb, lr, lhi);
      v16h b = frag_W(W, N, kb, nt, lane);
      WMMA_F16(acc, a, b);
    }
    float bi = bias[nt + lr];
#pragma unroll
    for (int v = 0; v < 8; ++v) {
      float o = acc[v] + bi;
      if (relu) o = fmaxf(o, 0.f);
      if (toVT) Yout[(nt + lr) * 128 + row0 + lhi * 8 + v] = (_Float16)o;
      else      Yout[(row0 + lhi * 8 + v) * ystride + nt + lr] = (_Float16)o;
    }
  }
  if (!toVT) {
    for (int c = N + lr; c < zero_to; c += 16) {
#pragma unroll
      for (int v = 0; v < 8; ++v)
        Yout[(row0 + lhi * 8 + v) * ystride + c] = (_Float16)0.f;
    }
  }
}

// Cross attention both directions (dir0: q=rows[0,64) k,v=rows[64,128); dir1 swapped).
// Q/K: f16 [128][32] zero-padded. VT: f16 [32][128] transposed. S: f16 [64][64].
__device__ __forceinline__ void attention(const _Float16* Q, const _Float16* K,
                                          const _Float16* VT, _Float16* S, float* OUTF,
                                          int dcols, int octiles, float scale,
                                          int wave, int lr, int lhi, int tid) {
  for (int dir = 0; dir < 2; ++dir) {
    const int qb = dir ? 64 : 0;
    const int kb = dir ? 0 : 64;
    // scores: 4x4 tiles, 2 per wave; feature dim padded to 32
#pragma unroll
    for (int i = 0; i < 2; ++i) {
      int t = wave * 2 + i;
      int qt = t >> 2, kt = t & 3;
      v16h a = frag_A(Q + qb * 32, 32, qt * 16, 0, lr, lhi);
      v16h b = ld_cat(K + (kb + kt * 16 + lr) * 32 + lhi * 16);  // implicit k^T, contiguous
      v8f acc = {};
      WMMA_F16(acc, a, b);
#pragma unroll
      for (int v = 0; v < 8; ++v)
        S[(qt * 16 + lhi * 8 + v) * 64 + kt * 16 + lr] = (_Float16)(acc[v] * scale);
    }
    __syncthreads();
    if (tid < 64) {  // row-wise softmax, f32 math, f16 storage
      _Float16* row = S + tid * 64;
      float mx = -1e30f;
      for (int j = 0; j < 64; ++j) mx = fmaxf(mx, (float)row[j]);
      float sum = 0.f;
      for (int j = 0; j < 64; ++j) { float e = __expf((float)row[j] - mx); sum += e; row[j] = (_Float16)e; }
      float inv = 1.f / sum;
      for (int j = 0; j < 64; ++j) row[j] = (_Float16)((float)row[j] * inv);
    }
    __syncthreads();
    int ntiles = 4 * octiles;
    if (wave < ntiles) {  // whole waves -> EXEC all-ones for WMMA
      int rt = (octiles == 2) ? (wave >> 1) : wave;
      int ct = (octiles == 2) ? (wave & 1) : 0;
      v8f acc = {};
#pragma unroll
      for (int kk = 0; kk < 64; kk += 32) {
        v16h a = frag_A(S, 64, rt * 16, kk, lr, lhi);
        v16h b = ld_cat(VT + (ct * 16 + lr) * 128 + kb + kk + lhi * 16);  // contiguous keys
        WMMA_F16(acc, a, b);
      }
#pragma unroll
      for (int v = 0; v < 8; ++v)
        OUTF[(qb + rt * 16 + lhi * 8 + v) * dcols + ct * 16 + lr] = acc[v];
    }
    __syncthreads();
  }
}

// Per-row LayerNorm over f32 [128][C] at smem base; rebuild f16 input [128][S] in place:
// [nodes(8) | normed(C) | zeros]. Optional residual add and f32 save of the C values.
__device__ __forceinline__ void ln_rebuild(const Params& P, int env, unsigned char* smem,
                                           int C, int S,
                                           const float* gL, const float* gH,
                                           const float* bL, const float* bH,
                                           bool withNodes, const float* resid, float* save,
                                           int tid) {
  float buf[32]; float nd[8];
#pragma unroll
  for (int j = 0; j < 8; ++j) nd[j] = 0.f;
  if (tid < 128) {
    const float* A = (const float*)smem;
    for (int j = 0; j < C; ++j) buf[j] = A[tid * C + j];
    if (withNodes) {
      const float* np_ = (tid < 64) ? (P.nL + (size_t)env * 64 * 8 + tid * 8)
                                    : (P.nH + (size_t)env * 64 * 8 + (tid - 64) * 8);
      for (int j = 0; j < 8; ++j) nd[j] = np_[j];
    }
    float mean = 0.f;
    for (int j = 0; j < C; ++j) mean += buf[j];
    mean /= (float)C;
    float var = 0.f;
    for (int j = 0; j < C; ++j) { float d = buf[j] - mean; var += d * d; }
    var /= (float)C;
    float inv = rsqrtf(var + 1e-5f);
    const float* g = (tid < 64) ? gL : gH;
    const float* b = (tid < 64) ? bL : bH;
    for (int j = 0; j < C; ++j) {
      float o = (buf[j] - mean) * inv * g[j] + b[j];
      if (resid) o += resid[tid * 16 + j];
      buf[j] = o;
    }
    if (save) for (int j = 0; j < C; ++j) save[tid * 16 + j] = buf[j];
  }
  __syncthreads();  // all f32 reads complete before in-place f16 writes
  if (tid < 128) {
    _Float16* Xo = (_Float16*)smem + tid * S;
    int c = 0;
    if (withNodes) for (int j = 0; j < 8; ++j) Xo[c++] = (_Float16)nd[j];
    for (int j = 0; j < C; ++j) Xo[c++] = (_Float16)buf[j];
    for (; c < S; ++c) Xo[c] = (_Float16)0.f;
  }
  __syncthreads();
}

// ---------------- fused main kernel: one workgroup per (b,c) environment ----------------
__global__ __launch_bounds__(256) void soap_main(Params P, const _Float16* __restrict__ ws,
                                                 float* __restrict__ out) {
  __shared__ __align__(32) unsigned char smem[65536];
  _Float16* const X  = (_Float16*)(smem);          // 24KB: x(128x96) / x1(128x64) / x2,out3(128x32) / f32 attn-out
  _Float16* const Qm = (_Float16*)(smem + SM_QKV);
  _Float16* const Km = (_Float16*)(smem + SM_QKV + 8192);
  _Float16* const Vt = (_Float16*)(smem + SM_QKV + 16384);  // transposed V [32][128]
  _Float16* const HS = (_Float16*)(smem + SM_HS);  // 8KB: MLP hidden OR attention scores
  float*    const O2 = (float*)(smem + SM_O2);     // 8KB: out2 f32 residual

  const int env  = blockIdx.x;
  const int tid  = threadIdx.x;
  const int wave = tid >> 5;
  const int lane = tid & 31;
  const int lr   = lane & 15;
  const int lhi  = lane >> 4;
  const int row0 = wave * 16;

  // Phase 0: stage x_nce (rows 0-63 = L, 64-127 = H), f32 -> f16, pad 72->96
  for (int i = tid; i < 128 * 96; i += 256) {
    int r = i / 96, c = i % 96;
    float v = 0.f;
    if (c < 72)
      v = (r < 64) ? P.Lnce[(size_t)env * 64 * 72 + r * 72 + c]
                   : P.Hnce[(size_t)env * 64 * 72 + (r - 64) * 72 + c];
    X[i] = (_Float16)v;
  }
  __syncthreads();

  // ---- level 1 (d=32) ----
  for (int m = 0; m < 3; ++m) {
    _Float16* Ym = (_Float16*)(smem + SM_QKV + m * 8192);
    mlp_f16(X, 96, 96, ws + OFF_A1W1 + m * 3072, P.a1[m][2], 32, HS, 32, 32,
            true, false, row0, lr, lhi, lane);
    __syncthreads();
    mlp_f16(HS, 32, 32, ws + OFF_A1W2 + m * 1024, P.a1[m][3], 32, Ym, 32, 32,
            false, (m == 2), row0, lr, lhi, lane);
    __syncthreads();
  }
  attention(Qm, Km, Vt, HS, (float*)smem, 32, 2, 0.17677669529663687f, wave, lr, lhi, tid);
  ln_rebuild(P, env, smem, 32, 64, P.ln1g[0], P.ln1g[1], P.ln1b[0], P.ln1b[1],
             true, nullptr, nullptr, tid);                       // -> x1 [128][64] (40 used)

  // ---- level 2 (d=16) ----
  for (int m = 0; m < 3; ++m) {
    _Float16* Ym = (_Float16*)(smem + SM_QKV + m * 8192);
    mlp_f16(X, 64, 64, ws + OFF_A2W1 + m * 1024, P.a2[m][2], 16, HS, 32, 32,
            true, false, row0, lr, lhi, lane);
    __syncthreads();
    mlp_f16(HS, 32, 32, ws + OFF_A2W2 + m * 512, P.a2[m][3], 16, Ym, 32, 32,
            false, (m == 2), row0, lr, lhi, lane);
    __syncthreads();
  }
  attention(Qm, Km, Vt, HS, (float*)smem, 16, 1, 0.25f, wave, lr, lhi, tid);
  ln_rebuild(P, env, smem, 16, 32, P.ln2g[0], P.ln2g[1], P.ln2b[0], P.ln2b[1],
             true, nullptr, O2, tid);                            // -> x2 [128][32], save out2

  // ---- level 3 (d=16, residual) ----
  for (int m = 0; m < 3; ++m) {
    _Float16* Ym = (_Float16*)(smem + SM_QKV + m * 8192);
    mlp_f16(X, 32, 32, ws + OFF_A3W1 + m * 512, P.a3[m][2], 16, HS, 32, 32,
            true, false, row0, lr, lhi, lane);
    __syncthreads();
    mlp_f16(HS, 32, 32, ws + OFF_A3W2 + m * 512, P.a3[m][3], 16, Ym, 32, 32,
            false, (m == 2), row0, lr, lhi, lane);
    __syncthreads();
  }
  attention(Qm, Km, Vt, HS, (float*)smem, 16, 1, 0.25f, wave, lr, lhi, tid);
  ln_rebuild(P, env, smem, 16, 32, P.ln3g[0], P.ln3g[1], P.ln3b[0], P.ln3b[1],
             false, O2, nullptr, tid);                           // -> out3 [128][32] (16 used)

  // ---- aggregator ----
  float* const SC = (float*)(smem + 8192);   // [128] scores
  float* const VL = (float*)(smem + 8704);   // [128][16] values
  mlp_f16(X, 32, 32, ws + OFF_SCW1, P.sc[2], 16, HS, 32, 32, true, false, row0, lr, lhi, lane);
  __syncthreads();
  {
    v16h a = frag_A(HS, 32, row0, 0, lr, lhi);
    v16h b = frag_W(ws + OFF_SCW2, 16, 0, 0, lane);
    v8f acc = {};
    WMMA_F16(acc, a, b);
    if (lr == 0) {
      float b2 = P.sc[3][0];
#pragma unroll
      for (int v = 0; v < 8; ++v) SC[row0 + lhi * 8 + v] = acc[v] + b2;
    }
  }
  __syncthreads();
  mlp_f16(X, 32, 32, ws + OFF_VLW1, P.vl[2], 16, HS, 32, 32, true, false, row0, lr, lhi, lane);
  __syncthreads();
  {
    v16h a = frag_A(HS, 32, row0, 0, lr, lhi);
    v16h b = frag_W(ws + OFF_VLW2, 16, 0, 0, lane);
    v8f acc = {};
    WMMA_F16(acc, a, b);
    float b2 = P.vl[3][lr];
#pragma unroll
    for (int v = 0; v < 8; ++v) VL[(row0 + lhi * 8 + v) * 16 + lr] = acc[v] + b2;
  }
  __syncthreads();
  if (tid < 16) {  // softmax over 128 nodes + weighted sum
    float mx = -1e30f;
    for (int e = 0; e < 128; ++e) mx = fmaxf(mx, SC[e]);
    float den = 0.f;
    for (int e = 0; e < 128; ++e) den += __expf(SC[e] - mx);
    float o = 0.f;
    for (int e = 0; e < 128; ++e) o += __expf(SC[e] - mx) * VL[e * 16 + tid];
    out[(size_t)env * 16 + tid] = o / den;
  }
}

// ---------------- host launch ----------------
extern "C" void kernel_launch(void* const* d_in, const int* in_sizes, int n_in,
                              void* d_out, int out_size, void* d_ws, size_t ws_size,
                              hipStream_t stream) {
  (void)in_sizes; (void)n_in; (void)out_size; (void)ws_size;
  auto F = [&](int i) { return (const float*)d_in[i]; };
  Params P;
  P.Lnce = F(0); P.Hnce = F(1); P.nL = F(2); P.nH = F(3);
  // params pytree, dicts sorted by key:
  // aggr{score{W1,W2,b1,b2}, val{...}}, attn1..3{k,q,v}{W1,W2,b1,b2}, normH1..3{b,g}, normL1..3{b,g}
  P.sc[0] = F(4);  P.sc[1] = F(5);  P.sc[2] = F(6);  P.sc[3] = F(7);
  P.vl[0] = F(8);  P.vl[1] = F(9);  P.vl[2] = F(10); P.vl[3] = F(11);
  for (int j = 0; j < 4; ++j) {
    P.a1[1][j] = F(12 + j); P.a1[0][j] = F(16 + j); P.a1[2][j] = F(20 + j);
    P.a2[1][j] = F(24 + j); P.a2[0][j] = F(28 + j); P.a2[2][j] = F(32 + j);
    P.a3[1][j] = F(36 + j); P.a3[0][j] = F(40 + j); P.a3[2][j] = F(44 + j);
  }
  P.ln1b[1] = F(48); P.ln1g[1] = F(49);
  P.ln2b[1] = F(50); P.ln2g[1] = F(51);
  P.ln3b[1] = F(52); P.ln3g[1] = F(53);
  P.ln1b[0] = F(54); P.ln1g[0] = F(55);
  P.ln2b[0] = F(56); P.ln2g[0] = F(57);
  P.ln3b[0] = F(58); P.ln3g[0] = F(59);

  _Float16* ws = (_Float16*)d_ws;
  soap_prep<<<22, 256, 0, stream>>>(P, ws);
  soap_main<<<4096, 256, 0, stream>>>(P, ws, (float*)d_out);
}
